// VectorQuantizer_26001732010068
// MI455X (gfx1250) — matrix operand order repaired and checked by reading, not compiled, not run on hardware
//
#include <hip/hip_runtime.h>
#include <hip/hip_bf16.h>

#define VQ_B 32768
#define VQ_D 512
#define VQ_K 8192
#define LDSB_STRIDE 520   // 512 halves + 8 pad -> conflict-free ds_load_b128

typedef __attribute__((ext_vector_type(16))) _Float16 v16h;
typedef __attribute__((ext_vector_type(8)))  _Float16 v8h;
typedef __attribute__((ext_vector_type(8)))  float    v8f;

// ---------------------------------------------------------------------------
// Kernel 1: convert codebook f32 -> f16 (workspace), compute ||w_k||^2,
//           and zero the loss accumulator. One wave per codebook row.
// ---------------------------------------------------------------------------
__global__ __launch_bounds__(256) void vq_prep(const float* __restrict__ cb,
                                               _Float16* __restrict__ cb16,
                                               float* __restrict__ cnorm,
                                               float* __restrict__ loss_acc) {
    const int t    = threadIdx.x;
    const int lane = t & 31;
    const int wid  = t >> 5;
    const int row  = blockIdx.x * 8 + wid;

    const float* p = cb + (size_t)row * VQ_D + lane * 16;
    float4 f0 = *(const float4*)(p + 0);
    float4 f1 = *(const float4*)(p + 4);
    float4 f2 = *(const float4*)(p + 8);
    float4 f3 = *(const float4*)(p + 12);

    float s = f0.x*f0.x + f0.y*f0.y + f0.z*f0.z + f0.w*f0.w
            + f1.x*f1.x + f1.y*f1.y + f1.z*f1.z + f1.w*f1.w
            + f2.x*f2.x + f2.y*f2.y + f2.z*f2.z + f2.w*f2.w
            + f3.x*f3.x + f3.y*f3.y + f3.z*f3.z + f3.w*f3.w;

    v8h h0, h1;
    h0[0]=(_Float16)f0.x; h0[1]=(_Float16)f0.y; h0[2]=(_Float16)f0.z; h0[3]=(_Float16)f0.w;
    h0[4]=(_Float16)f1.x; h0[5]=(_Float16)f1.y; h0[6]=(_Float16)f1.z; h0[7]=(_Float16)f1.w;
    h1[0]=(_Float16)f2.x; h1[1]=(_Float16)f2.y; h1[2]=(_Float16)f2.z; h1[3]=(_Float16)f2.w;
    h1[4]=(_Float16)f3.x; h1[5]=(_Float16)f3.y; h1[6]=(_Float16)f3.z; h1[7]=(_Float16)f3.w;

    _Float16* q = cb16 + (size_t)row * VQ_D + lane * 16;
    *(v8h*)(q)     = h0;
    *(v8h*)(q + 8) = h1;

    #pragma unroll
    for (int sft = 16; sft >= 1; sft >>= 1) s += __shfl_xor(s, sft, 32);
    if (lane == 0) cnorm[row] = s;
    if (row == 0 && t == 0) loss_acc[0] = 0.0f;
}

// ---------------------------------------------------------------------------
// Kernel 2: fused GEMM + argmin.
// Block = 256 threads = 8 waves, 256 rows of x per block (32 rows/wave:
// two register-resident A tiles). B (16 codebook rows, f16) streamed through
// double-buffered LDS; each B fragment feeds TWO WMMAs (acc0/acc1), halving
// DS bandwidth per WMMA and giving two independent accumulation chains.
// Score = cnorm[n] - 2 * dot(x_m, w_n); per-lane running argmin.
// ---------------------------------------------------------------------------
__global__ __launch_bounds__(256) void vq_main(const float* __restrict__ x,
                                               const _Float16* __restrict__ cb16,
                                               const float* __restrict__ cnorm,
                                               int* __restrict__ idx_out,
                                               float* __restrict__ idxf_out) {
    __shared__ _Float16 ldsB[2][16 * LDSB_STRIDE];

    const int t       = threadIdx.x;
    const int lane    = t & 31;
    const int wid     = t >> 5;
    const int hi      = lane >> 4;        // half-wave: 0 or 1
    const int c       = lane & 15;        // column / row-in-tile
    const int rowbase = blockIdx.x * 256 + wid * 32;

    // ---- Load A fragments: two 16-row tiles (rowbase..+15, rowbase+16..+31)
    // lanes 0-15:  halves 0..7 = K 0..7,  halves 8..15 = K 16..23
    // lanes 16-31: halves 0..7 = K 8..15, halves 8..15 = K 24..31
    const float* xr0 = x + (size_t)(rowbase + c) * VQ_D;
    const float* xr1 = xr0 + (size_t)16 * VQ_D;
    const int s = hi ? 8 : 0;
    v16h a0[16], a1[16];
    #pragma unroll
    for (int kk = 0; kk < 16; ++kk) {
        {
            const float* p0 = xr0 + kk * 32 + s;
            float4 f0 = *(const float4*)(p0);
            float4 f1 = *(const float4*)(p0 + 4);
            float4 g0 = *(const float4*)(p0 + 16);
            float4 g1 = *(const float4*)(p0 + 20);
            v16h av;
            av[0] =(_Float16)f0.x; av[1] =(_Float16)f0.y; av[2] =(_Float16)f0.z; av[3] =(_Float16)f0.w;
            av[4] =(_Float16)f1.x; av[5] =(_Float16)f1.y; av[6] =(_Float16)f1.z; av[7] =(_Float16)f1.w;
            av[8] =(_Float16)g0.x; av[9] =(_Float16)g0.y; av[10]=(_Float16)g0.z; av[11]=(_Float16)g0.w;
            av[12]=(_Float16)g1.x; av[13]=(_Float16)g1.y; av[14]=(_Float16)g1.z; av[15]=(_Float16)g1.w;
            a0[kk] = av;
        }
        {
            const float* p1 = xr1 + kk * 32 + s;
            float4 f0 = *(const float4*)(p1);
            float4 f1 = *(const float4*)(p1 + 4);
            float4 g0 = *(const float4*)(p1 + 16);
            float4 g1 = *(const float4*)(p1 + 20);
            v16h av;
            av[0] =(_Float16)f0.x; av[1] =(_Float16)f0.y; av[2] =(_Float16)f0.z; av[3] =(_Float16)f0.w;
            av[4] =(_Float16)f1.x; av[5] =(_Float16)f1.y; av[6] =(_Float16)f1.z; av[7] =(_Float16)f1.w;
            av[8] =(_Float16)g0.x; av[9] =(_Float16)g0.y; av[10]=(_Float16)g0.z; av[11]=(_Float16)g0.w;
            av[12]=(_Float16)g1.x; av[13]=(_Float16)g1.y; av[14]=(_Float16)g1.z; av[15]=(_Float16)g1.w;
            a1[kk] = av;
        }
    }

    // ---- Preload B tile 0 into LDS (16 rows x 512 halves, padded stride) ----
    {
        const uint4* src = (const uint4*)cb16;
        #pragma unroll
        for (int j = 0; j < 4; ++j) {
            int id = t + j * 256;                 // 1024 x 16B chunks
            uint4 v = src[id];
            *(uint4*)&ldsB[0][(id >> 6) * LDSB_STRIDE + (id & 63) * 8] = v;
        }
    }
    __syncthreads();

    float minv0[8], minv1[8];
    int   mini0[8], mini1[8];
    #pragma unroll
    for (int r = 0; r < 8; ++r) {
        minv0[r] = 3.4e38f; mini0[r] = 0;
        minv1[r] = 3.4e38f; mini1[r] = 0;
    }

    const int boff = hi ? 8 : 0;

    for (int i = 0; i < VQ_K / 16; ++i) {
        const int buf = i & 1;
        const bool has_next = (i + 1) < (VQ_K / 16);

        // prefetch next B tile into registers (latency hidden under WMMA)
        uint4 pf0, pf1, pf2, pf3;
        if (has_next) {
            const uint4* src = (const uint4*)(cb16 + (size_t)(i + 1) * 16 * VQ_D);
            pf0 = src[t];       pf1 = src[t + 256];
            pf2 = src[t + 512]; pf3 = src[t + 768];
        }
        float cn = cnorm[i * 16 + c];

        // ---- 16 B fragments, each feeding two chained WMMAs ----
        v8f acc0 = {};
        v8f acc1 = {};
        #pragma unroll
        for (int kk = 0; kk < 16; ++kk) {
            const _Float16* bp = &ldsB[buf][c * LDSB_STRIDE + kk * 32 + boff];
            v8h b0 = *(const v8h*)(bp);
            v8h b1 = *(const v8h*)(bp + 16);
            v16h bv;
            #pragma unroll
            for (int j = 0; j < 8; ++j) { bv[j] = b0[j]; bv[j + 8] = b1[j]; }
            acc0 = __builtin_amdgcn_wmma_f32_16x16x32_f16(
                false, a0[kk], false, bv, (short)0, acc0, false, false);
            acc1 = __builtin_amdgcn_wmma_f32_16x16x32_f16(
                false, a1[kk], false, bv, (short)0, acc1, false, false);
        }

        // ---- fold into running per-lane argmin ----
        #pragma unroll
        for (int r = 0; r < 8; ++r) {
            float d0 = __builtin_fmaf(acc0[r], -2.0f, cn);
            if (d0 < minv0[r]) { minv0[r] = d0; mini0[r] = i * 16 + c; }
            float d1 = __builtin_fmaf(acc1[r], -2.0f, cn);
            if (d1 < minv1[r]) { minv1[r] = d1; mini1[r] = i * 16 + c; }
        }

        __syncthreads();                       // everyone done reading buf^1 (iter i-1)
        if (has_next) {
            const int nb = buf ^ 1;
            int id0 = t;        *(uint4*)&ldsB[nb][(id0 >> 6) * LDSB_STRIDE + (id0 & 63) * 8] = pf0;
            int id1 = t + 256;  *(uint4*)&ldsB[nb][(id1 >> 6) * LDSB_STRIDE + (id1 & 63) * 8] = pf1;
            int id2 = t + 512;  *(uint4*)&ldsB[nb][(id2 >> 6) * LDSB_STRIDE + (id2 & 63) * 8] = pf2;
            int id3 = t + 768;  *(uint4*)&ldsB[nb][(id3 >> 6) * LDSB_STRIDE + (id3 & 63) * 8] = pf3;
        }
        __syncthreads();                       // stores visible before next compute
    }

    // ---- final cross-lane argmin within each 16-lane half ----
    #pragma unroll
    for (int r = 0; r < 8; ++r) {
        #pragma unroll
        for (int sft = 8; sft >= 1; sft >>= 1) {
            float ov0 = __shfl_xor(minv0[r], sft, 32);
            int   oi0 = __shfl_xor(mini0[r], sft, 32);
            if (ov0 < minv0[r] || (ov0 == minv0[r] && oi0 < mini0[r])) {
                minv0[r] = ov0; mini0[r] = oi0;
            }
            float ov1 = __shfl_xor(minv1[r], sft, 32);
            int   oi1 = __shfl_xor(mini1[r], sft, 32);
            if (ov1 < minv1[r] || (ov1 == minv1[r] && oi1 < mini1[r])) {
                minv1[r] = ov1; mini1[r] = oi1;
            }
        }
    }
    // lane 0 writes rows rb+0..7 of each tile; lane 16 writes rb+8..15
    if (c == 0) {
        const int rb0 = rowbase + (hi ? 8 : 0);
        const int rb1 = rowbase + 16 + (hi ? 8 : 0);
        #pragma unroll
        for (int r = 0; r < 8; ++r) {
            idx_out[rb0 + r]  = mini0[r];
            idxf_out[rb0 + r] = (float)mini0[r];
            idx_out[rb1 + r]  = mini1[r];
            idxf_out[rb1 + r] = (float)mini1[r];
        }
    }
}

// ---------------------------------------------------------------------------
// Kernel 3: gather quantized = codebook[idx], accumulate SSE for the loss.
// Block handles 4 rows (64 threads x 8 floats per row).
// ---------------------------------------------------------------------------
__global__ __launch_bounds__(256) void vq_gather(const float* __restrict__ x,
                                                 const float* __restrict__ cb,
                                                 const int* __restrict__ idx,
                                                 float* __restrict__ quant,
                                                 float* __restrict__ loss_acc) {
    const int t    = threadIdx.x;
    const int lane = t & 31;
    const int wid  = t >> 5;
    const int row  = blockIdx.x * 4 + (t >> 6);
    const int col  = (t & 63) * 8;

    const int k = idx[row];
    const float4* q4 = (const float4*)(cb + (size_t)k * VQ_D + col);
    const float4* x4 = (const float4*)(x + (size_t)row * VQ_D + col);
    float4 q0 = q4[0], q1 = q4[1];
    float4 x0 = x4[0], x1 = x4[1];

    float4* o4 = (float4*)(quant + (size_t)row * VQ_D + col);
    o4[0] = q0; o4[1] = q1;   // straight-through: x + (q - x) == q

    float dx, ss = 0.0f;
    dx = q0.x - x0.x; ss += dx * dx;
    dx = q0.y - x0.y; ss += dx * dx;
    dx = q0.z - x0.z; ss += dx * dx;
    dx = q0.w - x0.w; ss += dx * dx;
    dx = q1.x - x1.x; ss += dx * dx;
    dx = q1.y - x1.y; ss += dx * dx;
    dx = q1.z - x1.z; ss += dx * dx;
    dx = q1.w - x1.w; ss += dx * dx;

    #pragma unroll
    for (int sft = 16; sft >= 1; sft >>= 1) ss += __shfl_xor(ss, sft, 32);

    __shared__ float red[8];
    if (lane == 0) red[wid] = ss;
    __syncthreads();
    if (t == 0) {
        float tot = red[0] + red[1] + red[2] + red[3]
                  + red[4] + red[5] + red[6] + red[7];
        atomicAdd(loss_acc, tot);
    }
}

// loss = q_latent + 0.25 * e_latent = 1.25 * SSE / (B*D)
__global__ void vq_finalize(const float* __restrict__ loss_acc,
                            float* __restrict__ out_loss) {
    out_loss[0] = loss_acc[0] * (1.25f / ((float)VQ_B * (float)VQ_D));
}

// ---------------------------------------------------------------------------
extern "C" void kernel_launch(void* const* d_in, const int* in_sizes, int n_in,
                              void* d_out, int out_size, void* d_ws, size_t ws_size,
                              hipStream_t stream) {
    const float* x  = (const float*)d_in[0];   // [32768, 512] f32
    const float* cb = (const float*)d_in[1];   // [8192, 512]  f32

    float* out      = (float*)d_out;
    float* quant    = out;                               // B*D
    float* out_idx  = out + (size_t)VQ_B * VQ_D;         // B (indices as float)
    float* out_loss = out_idx + VQ_B;                    // 1

    char* ws = (char*)d_ws;
    _Float16* cb16  = (_Float16*)ws;                                       // 8 MB
    float*    cnorm = (float*)(ws + (size_t)VQ_K * VQ_D * 2);              // 32 KB
    int*      idx   = (int*)(ws + (size_t)VQ_K * VQ_D * 2 + VQ_K * 4);     // 128 KB
    float* loss_acc = (float*)(ws + (size_t)VQ_K * VQ_D * 2 + VQ_K * 4
                                  + (size_t)VQ_B * 4);

    vq_prep    <<<VQ_K / 8,   256, 0, stream>>>(cb, cb16, cnorm, loss_acc);
    vq_main    <<<VQ_B / 256, 256, 0, stream>>>(x, cb16, cnorm, idx, out_idx);
    vq_gather  <<<VQ_B / 4,   256, 0, stream>>>(x, cb, idx, quant, loss_acc);
    vq_finalize<<<1, 1, 0, stream>>>(loss_acc, out_loss);
}
